// LovaszSoftmaxLoss_14731737825634
// MI455X (gfx1250) — matrix operand order repaired and checked by reading, not compiled, not run on hardware
//
#include <hip/hip_runtime.h>
#include <hip/hip_bf16.h>

typedef __attribute__((ext_vector_type(16))) _Float16 v16h;
typedef __attribute__((ext_vector_type(8)))  float    v8f;

#define IGNORE_LBL 255
#define HW_IN   64
#define HW_OUT  512
#define NCLS    19
#define NB      2
#define NPLANE  (NB * NCLS)            // 38
#define PIX_IMG (HW_OUT * HW_OUT)      // 262144
#define NPIX    (NB * PIX_IMG)         // 524288 == 2^19
#define SORT_TILE 1024                 // elements per LDS sort tile (8 KB of u64)

// ---------------------------------------------------------------------------
// small utility kernels
// ---------------------------------------------------------------------------
__global__ void k_zero_f16(_Float16* p, int n) {
    int i = blockIdx.x * blockDim.x + threadIdx.x;
    if (i < n) p[i] = (_Float16)0.0f;
}
__global__ void k_zero_f32(float* p, int n) {
    int i = blockIdx.x * blockDim.x + threadIdx.x;
    if (i < n) p[i] = 0.0f;
}
__global__ void k_cvt_pred(const float* __restrict__ pred, _Float16* __restrict__ predH, int n) {
    int i = blockIdx.x * blockDim.x + threadIdx.x;
    if (i < n) predH[i] = (_Float16)pred[i];
}

// Build f16 interpolation matrices for align_corners bilinear 64 -> 512.
// WxT: [64 x 512]  (WxT[k][n] = weight of source col k for output col n)
// Wy : [512 x 64]  (Wy[n][k]  = weight of source row k for output row n)
__global__ void k_build_weights(_Float16* __restrict__ WxT, _Float16* __restrict__ Wy) {
    int n = blockIdx.x * blockDim.x + threadIdx.x;
    if (n >= HW_OUT) return;
    float xs = (float)n * ((float)(HW_IN - 1) / (float)(HW_OUT - 1));
    int   x0 = (int)floorf(xs);
    int   x1 = min(x0 + 1, HW_IN - 1);
    float w  = xs - (float)x0;
    WxT[x0 * HW_OUT + n] = (_Float16)(1.0f - w);
    if (x1 != x0) WxT[x1 * HW_OUT + n] = (_Float16)w;
    Wy[n * HW_IN + x0] = (_Float16)(1.0f - w);
    if (x1 != x0) Wy[n * HW_IN + x1] = (_Float16)w;
}

// ---------------------------------------------------------------------------
// WMMA 16x16 tile GEMM (f16 in, f32 acc) per CDNA5 fragment layouts
//   A 16x32 f16 : lanes 0-15 -> M, regs carry K (lane-half selects K subrange)
//   B 32x16 f16 : lanes 0-15 -> N, regs carry K (lane-half selects K subrange)
//   C/D 16x16 f32: lane = N (mod 16), reg v = M (+8 for lanes 16-31)
// ---------------------------------------------------------------------------
__device__ __forceinline__ void wmma_tile_f16(const _Float16* __restrict__ A, int lda,
                                              const _Float16* __restrict__ B, int ldb,
                                              int tileM, int tileN, int K, v8f& acc) {
    const int lane = threadIdx.x & 31;
    const int half = lane >> 4;   // 0 or 1
    const int l15  = lane & 15;
    for (int kk = 0; kk < K; kk += 32) {
        v16h af, bf;
#pragma unroll
        for (int j = 0; j < 16; ++j) {
            const int v = j >> 1, p = j & 1;
            const int ka = ((v < 4) ? 0 : 16) + half * 8 + (v & 3) * 2 + p;
            af[j] = A[(size_t)(tileM + l15) * lda + (kk + ka)];
        }
        const int kb = kk + 8 * half;
#pragma unroll
        for (int j = 0; j < 16; ++j) {
            bf[j] = B[(size_t)(kb + (j < 8 ? j : j + 8)) * ldb + (tileN + l15)];
        }
        acc = __builtin_amdgcn_wmma_f32_16x16x32_f16(false, af, false, bf,
                                                     (short)0, acc, false, false);
    }
}

// Step 1: T[q] (64x512, f16) = X[q] (64x64) * WxT (64x512)
__global__ void k_upsample_x(const _Float16* __restrict__ predH,
                             const _Float16* __restrict__ WxT,
                             _Float16* __restrict__ T) {
    const int q  = blockIdx.z;
    const int tm = blockIdx.y * 16;
    const int tn = blockIdx.x * 16;
    const _Float16* A = predH + (size_t)q * HW_IN * HW_IN;
    v8f acc = {};
    wmma_tile_f16(A, HW_IN, WxT, HW_OUT, tm, tn, HW_IN, acc);
    const int lane  = threadIdx.x & 31;
    const int n     = tn + (lane & 15);
    const int mbase = tm + 8 * (lane >> 4);
    _Float16* Out = T + (size_t)q * HW_IN * HW_OUT;
#pragma unroll
    for (int v = 0; v < 8; ++v)
        Out[(size_t)(mbase + v) * HW_OUT + n] = (_Float16)acc[v];
}

// Step 2: logits[q] (512x512, f32) = Wy (512x64) * T[q] (64x512)
__global__ void k_upsample_y(const _Float16* __restrict__ T,
                             const _Float16* __restrict__ Wy,
                             float* __restrict__ logits) {
    const int q  = blockIdx.z;
    const int tm = blockIdx.y * 16;
    const int tn = blockIdx.x * 16;
    const _Float16* B = T + (size_t)q * HW_IN * HW_OUT;
    v8f acc = {};
    wmma_tile_f16(Wy, HW_IN, B, HW_OUT, tm, tn, HW_IN, acc);
    const int lane  = threadIdx.x & 31;
    const int n     = tn + (lane & 15);
    const int mbase = tm + 8 * (lane >> 4);
    float* Out = logits + (size_t)q * PIX_IMG;
#pragma unroll
    for (int v = 0; v < 8; ++v)
        Out[(size_t)(mbase + v) * HW_OUT + n] = acc[v];
}

// ---------------------------------------------------------------------------
// Fused softmax / error-key construction / weighted-CE reduction / gts counts
// sortbuf key: hi32 = order-preserving uint of (valid ? err : -1), lo bits: fg|valid
// ---------------------------------------------------------------------------
__global__ void k_softmax_err_ce(const float* __restrict__ logits,
                                 const int* __restrict__ target,
                                 const float* __restrict__ cw,
                                 unsigned long long* __restrict__ sortbuf,
                                 float* __restrict__ gts,
                                 float* __restrict__ ce_acc /* [num, den] */) {
    __shared__ float sgts[NCLS];
    __shared__ float sred[256];
    const int t = threadIdx.x;
    if (t < NCLS) sgts[t] = 0.0f;
    __syncthreads();

    const int p = blockIdx.x * blockDim.x + t;
    float num = 0.0f, den = 0.0f;
    if (p < NPIX) {
        const int b  = p / PIX_IMG;
        const int hw = p - b * PIX_IMG;
        const float* lp = logits + (size_t)b * NCLS * PIX_IMG + hw;
        float lg[NCLS];
        float mx = -1e30f;
#pragma unroll
        for (int c = 0; c < NCLS; ++c) { lg[c] = lp[(size_t)c * PIX_IMG]; mx = fmaxf(mx, lg[c]); }
        float s = 0.0f;
#pragma unroll
        for (int c = 0; c < NCLS; ++c) { lg[c] = __expf(lg[c] - mx); s += lg[c]; }
        const float inv = 1.0f / s;

        const int  lbl   = target[p];
        const bool valid = (lbl != IGNORE_LBL);
        const int  ls    = valid ? lbl : 0;

        const float pr_l = lg[ls] * inv;
        const float nll  = -__logf(fmaxf(pr_l, 1e-30f));
        const float w    = valid ? cw[ls] : 0.0f;
        num = w * nll;
        den = w;

#pragma unroll
        for (int c = 0; c < NCLS; ++c) {
            const float pr  = lg[c] * inv;
            const float fg  = (valid && (c == ls)) ? 1.0f : 0.0f;
            const float err = fabsf(fg - pr);
            const float key = valid ? err : -1.0f;
            unsigned ku = __float_as_uint(key);
            ku = (ku & 0x80000000u) ? ~ku : (ku | 0x80000000u);
            const unsigned pay = (fg > 0.0f ? 1u : 0u) | (valid ? 2u : 0u);
            sortbuf[(size_t)c * NPIX + p] = ((unsigned long long)ku << 32) | pay;
            if (fg > 0.0f) atomicAdd(&sgts[c], 1.0f);
        }
    }

    sred[t] = num; __syncthreads();
    for (int off = 128; off > 0; off >>= 1) { if (t < off) sred[t] += sred[t + off]; __syncthreads(); }
    if (t == 0) atomicAdd(&ce_acc[0], sred[0]);
    __syncthreads();
    sred[t] = den; __syncthreads();
    for (int off = 128; off > 0; off >>= 1) { if (t < off) sred[t] += sred[t + off]; __syncthreads(); }
    if (t == 0) atomicAdd(&ce_acc[1], sred[0]);
    __syncthreads();
    if (t < NCLS && sgts[t] != 0.0f) atomicAdd(&gts[t], sgts[t]);
}

// ---------------------------------------------------------------------------
// Bitonic sort (descending final order). Direction flags always derive from the
// GLOBAL element index so LDS-tile stages compose with global merge stages.
// ---------------------------------------------------------------------------

// Global compare-exchange step (used only for j >= SORT_TILE)
__global__ void k_bitonic_global(unsigned long long* __restrict__ buf, int k, int j) {
    const unsigned i = blockIdx.x * blockDim.x + threadIdx.x;
    const unsigned l = i ^ (unsigned)j;
    unsigned long long* base = buf + (size_t)blockIdx.y * NPIX;
    if (l > i) {
        const unsigned long long a = base[i];
        const unsigned long long b = base[l];
        const bool desc = ((i & (unsigned)k) == 0u);
        if ((a < b) == desc) { base[i] = b; base[l] = a; }
    }
}

// LDS-resident compare-exchange over a SORT_TILE tile for one (k, j) stage.
// 512 threads, each owning one compare pair. i = pair-split index mapping.
__device__ __forceinline__ void tile_stage(unsigned long long* sbuf, unsigned gbase,
                                           unsigned t, int k, int j) {
    const unsigned i  = ((t & ~(unsigned)(j - 1)) << 1) | (t & (unsigned)(j - 1));
    const unsigned l  = i | (unsigned)j;
    const unsigned long long a = sbuf[i];
    const unsigned long long b = sbuf[l];
    const bool desc = (((gbase + i) & (unsigned)k) == 0u);
    if ((a < b) == desc) { sbuf[i] = b; sbuf[l] = a; }
}

// Full bitonic sort of each SORT_TILE tile: all stages k=2..SORT_TILE in LDS.
__global__ void k_bitonic_sort_local(unsigned long long* __restrict__ buf) {
    __shared__ unsigned long long sbuf[SORT_TILE];
    const unsigned t     = threadIdx.x;                 // 0..511
    const unsigned gbase = blockIdx.x * SORT_TILE;      // element offset in class
    unsigned long long* base = buf + (size_t)blockIdx.y * NPIX + gbase;
    sbuf[t]       = base[t];
    sbuf[t + 512] = base[t + 512];
    __syncthreads();
    for (int k = 2; k <= SORT_TILE; k <<= 1) {
        for (int j = k >> 1; j >= 1; j >>= 1) {
            tile_stage(sbuf, gbase, t, k, j);
            __syncthreads();
        }
    }
    base[t]       = sbuf[t];
    base[t + 512] = sbuf[t + 512];
}

// For k > SORT_TILE: run the remaining stages j = SORT_TILE/2 .. 1 in LDS.
__global__ void k_bitonic_merge_local(unsigned long long* __restrict__ buf, int k) {
    __shared__ unsigned long long sbuf[SORT_TILE];
    const unsigned t     = threadIdx.x;
    const unsigned gbase = blockIdx.x * SORT_TILE;
    unsigned long long* base = buf + (size_t)blockIdx.y * NPIX + gbase;
    sbuf[t]       = base[t];
    sbuf[t + 512] = base[t + 512];
    __syncthreads();
    for (int j = SORT_TILE >> 1; j >= 1; j >>= 1) {
        tile_stage(sbuf, gbase, t, k, j);
        __syncthreads();
    }
    base[t]       = sbuf[t];
    base[t + 512] = sbuf[t + 512];
}

// ---------------------------------------------------------------------------
// Lovasz gradient scan: one 1024-thread block per class, chunked inclusive scan
// ---------------------------------------------------------------------------
__global__ void k_lovasz(const unsigned long long* __restrict__ sortbuf,
                         const float* __restrict__ gts,
                         float* __restrict__ iou_acc) {
    const int c = blockIdx.x;
    const int t = threadIdx.x;
    __shared__ float sfg[1024];
    __shared__ float sval[1024];
    __shared__ float sbase[2];
    if (t == 0) { sbase[0] = 0.0f; sbase[1] = 0.0f; }

    const unsigned long long* base = sortbuf + (size_t)c * NPIX;
    const float gc = gts[c];
    float acc = 0.0f;

    for (int chunk = 0; chunk < NPIX; chunk += 1024) {
        const unsigned long long e = base[chunk + t];
        const unsigned pay = (unsigned)(e & 0xFFFFFFFFull);
        const unsigned ku  = (unsigned)(e >> 32);
        const unsigned fu  = (ku & 0x80000000u) ? (ku ^ 0x80000000u) : ~ku;
        float err = __uint_as_float(fu);
        const float fg  = (pay & 1u) ? 1.0f : 0.0f;
        const float val = (pay & 2u) ? 1.0f : 0.0f;
        err = (pay & 2u) ? err : 0.0f;   // err_s is zero for ignored pixels

        __syncthreads();
        sfg[t] = fg; sval[t] = val;
        __syncthreads();
        for (int off = 1; off < 1024; off <<= 1) {
            const float af = (t >= off) ? sfg[t - off]  : 0.0f;
            const float av = (t >= off) ? sval[t - off] : 0.0f;
            __syncthreads();
            sfg[t] += af; sval[t] += av;
            __syncthreads();
        }
        const float cfg  = sbase[0] + sfg[t];
        const float cval = sbase[1] + sval[t];

        const float uni  = gc + (cval - cfg);
        const float jac  = (uni > 0.0f) ? (1.0f - (gc - cfg) / fmaxf(uni, 1.0f)) : 1.0f;
        const float pfg  = cfg - fg, pval = cval - val;
        const float puni = gc + (pval - pfg);
        float pjac = (puni > 0.0f) ? (1.0f - (gc - pfg) / fmaxf(puni, 1.0f)) : 1.0f;
        if (chunk == 0 && t == 0) pjac = 0.0f;   // grad[0] = jacc[0]
        acc += err * (jac - pjac);

        __syncthreads();
        if (t == 1023) { sbase[0] = cfg; sbase[1] = cval; }
    }

    __syncthreads();
    sfg[t] = acc;
    __syncthreads();
    for (int off = 512; off > 0; off >>= 1) { if (t < off) sfg[t] += sfg[t + off]; __syncthreads(); }
    if (t == 0 && gc > 0.0f) atomicAdd(iou_acc, sfg[0]);
}

__global__ void k_finalize(const float* __restrict__ gts,
                           const float* __restrict__ iou_acc,
                           const float* __restrict__ ce_acc,
                           float* __restrict__ out) {
    int present = 0;
    for (int c = 0; c < NCLS; ++c) present += (gts[c] > 0.0f) ? 1 : 0;
    const float iou = iou_acc[0] / (float)max(present, 1);
    const float ce  = ce_acc[0] / ce_acc[1];
    out[0] = iou + ce;
}

// ---------------------------------------------------------------------------
extern "C" void kernel_launch(void* const* d_in, const int* in_sizes, int n_in,
                              void* d_out, int out_size, void* d_ws, size_t ws_size,
                              hipStream_t stream) {
    (void)in_sizes; (void)n_in; (void)out_size; (void)ws_size;
    const float* pred   = (const float*)d_in[0];
    const int*   target = (const int*)d_in[1];
    const float* cw     = (const float*)d_in[2];
    float*       out    = (float*)d_out;

    // workspace layout (~123 MB)
    char* ws = (char*)d_ws;
    size_t off = 0;
    _Float16* predH = (_Float16*)(ws + off); off += (size_t)NPLANE * HW_IN * HW_IN * 2;
    _Float16* WxT   = (_Float16*)(ws + off); off += (size_t)HW_IN * HW_OUT * 2;
    _Float16* Wy    = (_Float16*)(ws + off); off += (size_t)HW_OUT * HW_IN * 2;
    _Float16* T     = (_Float16*)(ws + off); off += (size_t)NPLANE * HW_IN * HW_OUT * 2;
    off = (off + 255) & ~(size_t)255;
    float* logits   = (float*)(ws + off);    off += (size_t)NPLANE * PIX_IMG * 4;
    off = (off + 255) & ~(size_t)255;
    unsigned long long* sortbuf = (unsigned long long*)(ws + off); off += (size_t)NCLS * NPIX * 8;
    float* stats    = (float*)(ws + off);    off += 32 * 4;
    float* gts      = stats;        // 19
    float* iou_acc  = stats + 19;   // 1
    float* ce_acc   = stats + 20;   // 2

    // init
    k_zero_f16<<<(HW_IN * HW_OUT + 255) / 256, 256, 0, stream>>>(WxT, HW_IN * HW_OUT);
    k_zero_f16<<<(HW_OUT * HW_IN + 255) / 256, 256, 0, stream>>>(Wy, HW_OUT * HW_IN);
    k_zero_f32<<<1, 32, 0, stream>>>(stats, 22);
    k_build_weights<<<2, 256, 0, stream>>>(WxT, Wy);
    k_cvt_pred<<<(NPLANE * HW_IN * HW_IN + 255) / 256, 256, 0, stream>>>(
        pred, predH, NPLANE * HW_IN * HW_IN);

    // bilinear upsample as two WMMA GEMMs
    k_upsample_x<<<dim3(HW_OUT / 16, HW_IN / 16, NPLANE), 32, 0, stream>>>(predH, WxT, T);
    k_upsample_y<<<dim3(HW_OUT / 16, HW_OUT / 16, NPLANE), 32, 0, stream>>>(T, Wy, logits);

    // softmax + error keys + CE
    k_softmax_err_ce<<<NPIX / 256, 256, 0, stream>>>(logits, target, cw, sortbuf, gts, ce_acc);

    // per-class descending bitonic sort (NPIX = 2^19), LDS tiles for j < SORT_TILE
    k_bitonic_sort_local<<<dim3(NPIX / SORT_TILE, NCLS), 512, 0, stream>>>(sortbuf);
    for (int k = 2 * SORT_TILE; k <= NPIX; k <<= 1) {
        for (int j = k >> 1; j >= SORT_TILE; j >>= 1)
            k_bitonic_global<<<dim3(NPIX / 256, NCLS), 256, 0, stream>>>(sortbuf, k, j);
        k_bitonic_merge_local<<<dim3(NPIX / SORT_TILE, NCLS), 512, 0, stream>>>(sortbuf, k);
    }

    // Lovasz scan + combine
    k_lovasz<<<NCLS, 1024, 0, stream>>>(sortbuf, gts, iou_acc);
    k_finalize<<<1, 1, 0, stream>>>(gts, iou_acc, ce_acc, out);
}
